// SwinTransformer_67628555043014
// MI455X (gfx1250) — matrix operand order repaired and checked by reading, not compiled, not run on hardware
//
#include <hip/hip_runtime.h>

typedef __bf16 bf16;
typedef __attribute__((ext_vector_type(16))) __bf16 v16bf;
typedef __attribute__((ext_vector_type(8)))  __bf16 v8bf;
typedef __attribute__((ext_vector_type(4)))  __bf16 v4bf;
typedef __attribute__((ext_vector_type(8)))  float  v8f;

#define WS7 7
#define NW  49   // tokens per 7x7 window

// Build a 16-wide bf16 fragment from two contiguous 16-byte runs (b128 loads).
__device__ __forceinline__ v16bf frag2(const bf16* p0, const bf16* p1) {
  v8bf a = *(const v8bf*)p0;
  v8bf b = *(const v8bf*)p1;
  return __builtin_shufflevector(a, b, 0, 1, 2, 3, 4, 5, 6, 7,
                                 8, 9, 10, 11, 12, 13, 14, 15);
}

// ---------------------------------------------------------------------------
// fp32 [K,N] -> bf16 transposed [N,K] (weights, once per launch)
// ---------------------------------------------------------------------------
__global__ void f2bf_t_kernel(const float* __restrict__ in, bf16* __restrict__ out,
                              int K, int N) {
  long i = (long)blockIdx.x * blockDim.x + threadIdx.x;
  if (i >= (long)K * N) return;
  int n = (int)(i % N);
  int k = (int)(i / N);
  out[(long)n * K + k] = (bf16)in[i];
}

// ---------------------------------------------------------------------------
// Patch embed: 2x2 stride-2 conv, NCHW in -> token layout [B, H*W, 96] fp32
// ---------------------------------------------------------------------------
__global__ void patch_embed_kernel(const float* __restrict__ img,
                                   const float* __restrict__ wconv,
                                   const float* __restrict__ bias,
                                   float* __restrict__ out) {
  const int C = 96, H = 128, W = 128, IMG = 256;
  long idx = (long)blockIdx.x * blockDim.x + threadIdx.x;
  long total = (long)8 * H * W * C;
  if (idx >= total) return;
  int co = (int)(idx % C);
  long t = idx / C;
  int wx = (int)(t % W); t /= W;
  int hy = (int)(t % H);
  int b  = (int)(t / H);
  float acc = bias[co];
#pragma unroll
  for (int ci = 0; ci < 4; ci++)
#pragma unroll
    for (int kh = 0; kh < 2; kh++)
#pragma unroll
      for (int kw = 0; kw < 2; kw++)
        acc += img[(((long)b * 4 + ci) * IMG + (hy * 2 + kh)) * IMG + (wx * 2 + kw)] *
               wconv[((co * 4 + ci) * 2 + kh) * 2 + kw];
  out[idx] = acc;
}

// ---------------------------------------------------------------------------
// LayerNorm over last dim C, one wave per token. Optional fp32 / bf16 outputs.
// ---------------------------------------------------------------------------
__global__ void layernorm_kernel(const float* __restrict__ in, const float* __restrict__ g,
                                 const float* __restrict__ b, float* __restrict__ outF,
                                 bf16* __restrict__ outB, int C) {
  long t = blockIdx.x;
  const float* x = in + t * (long)C;
  int lane = threadIdx.x;
  float s = 0.f, s2 = 0.f;
  for (int c = lane; c < C; c += 32) { float v = x[c]; s += v; s2 += v * v; }
#pragma unroll
  for (int m = 16; m >= 1; m >>= 1) { s += __shfl_xor(s, m, 32); s2 += __shfl_xor(s2, m, 32); }
  float mu  = s / C;
  float inv = rsqrtf(s2 / C - mu * mu + 1e-5f);
  for (int c = lane; c < C; c += 32) {
    float v = (x[c] - mu) * inv * g[c] + b[c];
    if (outF) outF[t * (long)C + c] = v;
    if (outB) outB[t * (long)C + c] = (bf16)v;
  }
}

// ---------------------------------------------------------------------------
// Patch-merge 2x2 gather: x [B,H,W,C] -> out [B,(H/2)(W/2),4C] fp32
// ---------------------------------------------------------------------------
__global__ void merge_gather_kernel(const float* __restrict__ x, float* __restrict__ out,
                                    int H, int W, int C) {
  int H2 = H >> 1, W2 = W >> 1, C4 = 4 * C;
  long idx = (long)blockIdx.x * blockDim.x + threadIdx.x;
  long total = (long)8 * H2 * W2 * C4;
  if (idx >= total) return;
  int cc = (int)(idx % C4);
  long t = idx / C4;
  int w2 = (int)(t % W2); t /= W2;
  int h2 = (int)(t % H2);
  int b  = (int)(t / H2);
  int q = cc / C, c = cc % C;
  int dh = (q == 1 || q == 3) ? 1 : 0;
  int dw = (q >= 2) ? 1 : 0;
  out[idx] = x[(((long)b * H + (2 * h2 + dh)) * W + (2 * w2 + dw)) * C + c];
}

// ---------------------------------------------------------------------------
// GEMM: C[M,N] = A[M,K](bf16, row-major) x Bt[N,K](bf16, pre-transposed)
//       (+bias)(+residual fp32).  One wave computes a 32x32 macro-tile
//       (2x2 WMMA tiles) so each fragment load feeds two WMMAs.
// K multiple of 32; M,N multiples of 32. All fragment loads are b128 pairs.
// ---------------------------------------------------------------------------
__global__ void gemm_bf16_wmma(const bf16* __restrict__ A, const bf16* __restrict__ Bt,
                               const float* __restrict__ bias, const float* __restrict__ res,
                               float* __restrict__ outF, bf16* __restrict__ outB,
                               int M, int N, int K) {
  int lane = threadIdx.x;
  int l15 = lane & 15, half = lane >> 4;
  long m0 = (long)blockIdx.y * 32;
  int  n0 = blockIdx.x * 32;
  const bf16* Ar0 = A + (m0 + l15) * (long)K;          // rows m0+l15, m0+16+l15
  const bf16* Ar1 = Ar0 + 16 * (long)K;
  const bf16* Br0 = Bt + (long)(n0 + l15) * K;         // cols n0+l15, n0+16+l15
  const bf16* Br1 = Br0 + 16 * (long)K;
  v8f acc00 = {}, acc01 = {}, acc10 = {}, acc11 = {};
  for (int k0 = 0; k0 < K; k0 += 32) {
    // A fragment: K = 8h..8h+7 and 16+8h..16+8h+7
    v16bf a0 = frag2(Ar0 + k0 + half * 8, Ar0 + k0 + 16 + half * 8);
    v16bf a1 = frag2(Ar1 + k0 + half * 8, Ar1 + k0 + 16 + half * 8);
    // B fragment: K = 8h..8h+7 and 16+8h..16+8h+7
    v16bf b0 = frag2(Br0 + k0 + half * 8, Br0 + k0 + 16 + half * 8);
    v16bf b1 = frag2(Br1 + k0 + half * 8, Br1 + k0 + 16 + half * 8);
    acc00 = __builtin_amdgcn_wmma_f32_16x16x32_bf16(false, a0, false, b0, (short)0, acc00, false, false);
    acc01 = __builtin_amdgcn_wmma_f32_16x16x32_bf16(false, a0, false, b1, (short)0, acc01, false, false);
    acc10 = __builtin_amdgcn_wmma_f32_16x16x32_bf16(false, a1, false, b0, (short)0, acc10, false, false);
    acc11 = __builtin_amdgcn_wmma_f32_16x16x32_bf16(false, a1, false, b1, (short)0, acc11, false, false);
  }
  auto stile = [&](const v8f& acc, int mt, int nt) {
    int n = n0 + nt * 16 + l15;
    float bv = bias ? bias[n] : 0.f;
#pragma unroll
    for (int j = 0; j < 8; j++) {                      // C/D: M = j + 8h, N = l15
      long m = m0 + mt * 16 + j + half * 8;
      float v = acc[j] + bv;
      if (res)  v += res[m * (long)N + n];
      if (outF) outF[m * (long)N + n] = v;
      if (outB) outB[m * (long)N + n] = (bf16)v;
    }
  };
  stile(acc00, 0, 0); stile(acc01, 0, 1);
  stile(acc10, 1, 0); stile(acc11, 1, 1);
}

// ---------------------------------------------------------------------------
// Windowed (shifted) multi-head attention for one (batch, window, head).
// 4 waves / block. Q,K in LDS row-major [64,32]; V transposed [32,64] so all
// WMMA fragments are contiguous b128 pairs. S = Q K^T (16 WMMAs), register
// softmax with 16-lane shfl reductions + arithmetic rel-pos bias + SW-MSA
// region mask, P via LDS, O = P V (4 WMMAs).
// ---------------------------------------------------------------------------
__global__ void attn_kernel(const bf16* __restrict__ qkv, const float* __restrict__ qkv_bias,
                            const float* __restrict__ bias_table, bf16* __restrict__ outB,
                            int H, int W, int C, int heads, int shift, int nWx) {
  int win  = blockIdx.x;
  int head = blockIdx.y;
  int b    = blockIdx.z;
  int wy = win / nWx, wx = win % nWx;
  int Hp = H + (WS7 - H % WS7) % WS7;
  int Wp = W + (WS7 - W % WS7) % WS7;
  int L  = H * W;
  int C3 = 3 * C;
  int co = head * 32;
  int tid = threadIdx.x;           // 0..127
  int wave = tid >> 5, lane = tid & 31;
  int l15 = lane & 15, half = lane >> 4;

  __shared__ bf16 Qs[64 * 32];
  __shared__ bf16 Ks[64 * 32];
  __shared__ bf16 Vt[32 * 64];     // V transposed: Vt[c][r]
  __shared__ bf16 Ps[64 * 64];
  __shared__ int  toks[64];
  __shared__ int  regs[64];

  if (tid < 64) {
    int r = tid;
    int tok = -1, rg = 9;
    if (r < NW) {
      int i = r / WS7, j = r % WS7;
      int hp = wy * WS7 + i, wp = wx * WS7 + j;
      int rh = hp < Hp - WS7 ? 0 : (hp < Hp - shift ? 1 : 2);
      int rw = wp < Wp - WS7 ? 0 : (wp < Wp - shift ? 1 : 2);
      rg = rh * 3 + rw;
      int h = hp + shift; if (h >= Hp) h -= Hp;   // undo roll(-shift)
      int w = wp + shift; if (w >= Wp) w -= Wp;
      tok = (h < H && w < W) ? (h * W + w) : -2;  // -2: zero-padded token
    }
    toks[r] = tok; regs[r] = rg;
  }
  __syncthreads();

  long baseB = (long)b * L;
  // gather Q/K/V for the window: 64 rows x 32 cols, 4 bf16 per thread-chunk
  for (int ch = tid; ch < 64 * 8; ch += 128) {
    int r = ch >> 3, c = (ch & 7) * 4;
    int tok = toks[r];
    v4bf qv = {}, kv = {}, vv = {};
    if (tok >= 0) {
      const bf16* p = qkv + (baseB + tok) * (long)C3 + co + c;
      qv = *(const v4bf*)(p);
      kv = *(const v4bf*)(p + C);
      vv = *(const v4bf*)(p + 2 * C);
    } else if (tok == -2) {                       // pad token: qkv = bias
#pragma unroll
      for (int i = 0; i < 4; i++) {
        kv[i] = (bf16)qkv_bias[C + co + c + i];
        vv[i] = (bf16)qkv_bias[2 * C + co + c + i];
      }
    }
    *(v4bf*)(Qs + r * 32 + c) = qv;
    *(v4bf*)(Ks + r * 32 + c) = kv;
#pragma unroll
    for (int i = 0; i < 4; i++) Vt[(c + i) * 64 + r] = vv[i];
  }
  __syncthreads();

  // ---- S = Q K^T : this wave owns rows [16*wave, 16*wave+16) -------------
  const bf16* Ar = Qs + (wave * 16 + l15) * 32;
  v16bf qf = frag2(Ar + half * 8, Ar + 16 + half * 8);
  v8f sacc[4];
#pragma unroll
  for (int t = 0; t < 4; t++) {
    const bf16* Kr = Ks + (t * 16 + l15) * 32;    // B[k][n]=K[n][k]: contiguous k
    v16bf kf = frag2(Kr + half * 8, Kr + 16 + half * 8);
    v8f z = {};
    sacc[t] = __builtin_amdgcn_wmma_f32_16x16x32_bf16(false, qf, false, kf,
                                                      (short)0, z, false, false);
  }

  // ---- logits: scale + rel-pos bias + shift mask -------------------------
  const float scale = 0.17677669529663687f;       // 32^-0.5
#pragma unroll
  for (int t = 0; t < 4; t++) {
    int n = t * 16 + l15;
#pragma unroll
    for (int j = 0; j < 8; j++) {
      int m = wave * 16 + j + half * 8;
      float v = sacc[t][j] * scale;
      if (m < NW && n < NW) {
        int ri = m / WS7, ci = m - ri * WS7;
        int rj = n / WS7, cj = n - rj * WS7;
        int bidx = (ri - rj + 6) * 13 + (ci - cj + 6);
        v += bias_table[bidx * heads + head];
        if (shift > 0 && regs[m] != regs[n]) v += -100.f;
      }
      if (n >= NW) v = -1e30f;                    // non-existent key columns
      sacc[t][j] = v;
    }
  }

  // ---- softmax over 64 cols (16-lane groups hold one row per tile) -------
  float rinv[8];
#pragma unroll
  for (int j = 0; j < 8; j++) {
    float mx = fmaxf(fmaxf(sacc[0][j], sacc[1][j]), fmaxf(sacc[2][j], sacc[3][j]));
    mx = fmaxf(mx, __shfl_xor(mx, 1, 32));
    mx = fmaxf(mx, __shfl_xor(mx, 2, 32));
    mx = fmaxf(mx, __shfl_xor(mx, 4, 32));
    mx = fmaxf(mx, __shfl_xor(mx, 8, 32));
    float s = 0.f;
#pragma unroll
    for (int t = 0; t < 4; t++) { float e = __expf(sacc[t][j] - mx); sacc[t][j] = e; s += e; }
    s += __shfl_xor(s, 1, 32);
    s += __shfl_xor(s, 2, 32);
    s += __shfl_xor(s, 4, 32);
    s += __shfl_xor(s, 8, 32);
    rinv[j] = 1.f / s;
  }
#pragma unroll
  for (int t = 0; t < 4; t++) {
    int n = t * 16 + l15;
#pragma unroll
    for (int j = 0; j < 8; j++) {
      int m = wave * 16 + j + half * 8;
      Ps[m * 64 + n] = (bf16)(sacc[t][j] * rinv[j]);
    }
  }
  __syncthreads();

  // ---- O = P V : rows [16*wave ..), 32 cols, K=64 ------------------------
  v8f oacc[2] = {};
#pragma unroll
  for (int kk = 0; kk < 2; kk++) {
    const bf16* Pr = Ps + (wave * 16 + l15) * 64 + kk * 32;
    v16bf pf = frag2(Pr + half * 8, Pr + 16 + half * 8);
#pragma unroll
    for (int t = 0; t < 2; t++) {
      const bf16* Vr = Vt + (t * 16 + l15) * 64 + kk * 32;  // B[k][n]=V[k][n]
      v16bf vf = frag2(Vr + half * 8, Vr + 16 + half * 8);
      oacc[t] = __builtin_amdgcn_wmma_f32_16x16x32_bf16(false, pf, false, vf,
                                                        (short)0, oacc[t], false, false);
    }
  }
#pragma unroll
  for (int t = 0; t < 2; t++) {
    int c = co + t * 16 + l15;
#pragma unroll
    for (int j = 0; j < 8; j++) {
      int m = wave * 16 + j + half * 8;
      if (m < NW) {
        int tok = toks[m];
        if (tok >= 0) outB[(baseB + tok) * (long)C + c] = (bf16)oacc[t][j];
      }
    }
  }
}

// ---------------------------------------------------------------------------
// Host orchestration
// ---------------------------------------------------------------------------
extern "C" void kernel_launch(void* const* d_in, const int* in_sizes, int n_in,
                              void* d_out, int out_size, void* d_ws, size_t ws_size,
                              hipStream_t stream) {
  (void)in_sizes; (void)out_size; (void)ws_size;
  if (n_in < 53) return;

  const float* xin  = (const float*)d_in[0];
  const float* pew  = (const float*)d_in[1];
  const float* peb  = (const float*)d_in[2];
  const float* peng = (const float*)d_in[3];
  const float* penb = (const float*)d_in[4];

  const int DIMS_[3]  = {96, 192, 384};
  const int HEADS_[3] = {3, 6, 12};

  const float *ng[3][2], *nb[3][2], *qw[3][2], *qb[3][2], *bt[3][2], *pw[3][2], *pb[3][2];
  const float *mng[2], *mnb[2], *mrw[2];
  int idx = 5;
  for (int li = 0; li < 3; li++) {
    for (int bi = 0; bi < 2; bi++) {
      ng[li][bi] = (const float*)d_in[idx++];
      nb[li][bi] = (const float*)d_in[idx++];
      qw[li][bi] = (const float*)d_in[idx++];
      qb[li][bi] = (const float*)d_in[idx++];
      bt[li][bi] = (const float*)d_in[idx++];
      pw[li][bi] = (const float*)d_in[idx++];
      pb[li][bi] = (const float*)d_in[idx++];
    }
    if (li < 2) {
      mng[li] = (const float*)d_in[idx++];
      mnb[li] = (const float*)d_in[idx++];
      mrw[li] = (const float*)d_in[idx++];
    }
  }

  // ---- workspace carve-up (time-sliced aliases keep footprint ~155MB) ----
  char* ws = (char*)d_ws;
  size_t off = 0;
  auto carve = [&](size_t bytes) -> void* {
    void* p = ws + off;
    off += (bytes + 255) & ~(size_t)255;
    return p;
  };
  const long Bn = 8;
  const long T1 = Bn * 128 * 128;                       // max token count

  float* xbuf  = (float*)carve((size_t)T1 * 96 * sizeof(float));    // residual stream
  void*  shrd  = carve((size_t)T1 * 288 * sizeof(bf16));            // union buffer
  float* tmpf  = (float*)shrd;                                      // conv-out / merged tokens
  bf16*  qkvbf = (bf16*)shrd;                                       // qkv activations
  bf16*  xnbf  = (bf16*)carve((size_t)T1 * 96 * sizeof(bf16));      // LN-out bf16
  bf16*  awbf  = xnbf;                                              // attn-out (aliases, sequential)

  bf16* qwb[3][2]; bf16* pwb[3][2]; bf16* rwb[2];
  for (int li = 0; li < 3; li++) {
    int C = DIMS_[li];
    for (int bi = 0; bi < 2; bi++) {
      qwb[li][bi] = (bf16*)carve((size_t)C * 3 * C * sizeof(bf16));
      pwb[li][bi] = (bf16*)carve((size_t)C * C * sizeof(bf16));
    }
    if (li < 2) rwb[li] = (bf16*)carve((size_t)4 * C * 2 * C * sizeof(bf16));
  }

  // weights: fp32 [K,N] -> bf16 transposed [N,K]
  auto cvt = [&](const float* src, bf16* dst, int K, int N) {
    long n = (long)K * N;
    f2bf_t_kernel<<<(unsigned)((n + 255) / 256), 256, 0, stream>>>(src, dst, K, N);
  };
  for (int li = 0; li < 3; li++) {
    int C = DIMS_[li];
    for (int bi = 0; bi < 2; bi++) {
      cvt(qw[li][bi], qwb[li][bi], C, 3 * C);
      cvt(pw[li][bi], pwb[li][bi], C, C);
    }
    if (li < 2) cvt(mrw[li], rwb[li], 4 * C, 2 * C);
  }

  // ---- patch embed + embed LN (fp32 residual stream) ---------------------
  {
    long total = T1 * 96;
    patch_embed_kernel<<<(unsigned)((total + 255) / 256), 256, 0, stream>>>(xin, pew, peb, tmpf);
    layernorm_kernel<<<(unsigned)T1, 32, 0, stream>>>(tmpf, peng, penb, xbuf, nullptr, 96);
  }

  int H = 128, W = 128;
  for (int li = 0; li < 3; li++) {
    int C = DIMS_[li], nh = HEADS_[li];
    long Lt = Bn * H * W;
    for (int bi = 0; bi < 2; bi++) {
      int shift = bi ? (WS7 / 2) : 0;
      // LN -> bf16
      layernorm_kernel<<<(unsigned)Lt, 32, 0, stream>>>(xbuf, ng[li][bi], nb[li][bi],
                                                        nullptr, xnbf, C);
      // QKV GEMM: [Lt,C] x [C,3C] + bias -> bf16
      dim3 g1((unsigned)(3 * C / 32), (unsigned)(Lt / 32));
      gemm_bf16_wmma<<<g1, 32, 0, stream>>>(xnbf, qwb[li][bi], qb[li][bi], nullptr,
                                            nullptr, qkvbf, (int)Lt, 3 * C, C);
      // windowed attention
      int Hp = H + (WS7 - H % WS7) % WS7;
      int Wp = W + (WS7 - W % WS7) % WS7;
      int nWx = Wp / WS7, nWy = Hp / WS7;
      dim3 ga((unsigned)(nWy * nWx), (unsigned)nh, (unsigned)Bn);
      attn_kernel<<<ga, 128, 0, stream>>>(qkvbf, qb[li][bi], bt[li][bi], awbf,
                                          H, W, C, nh, shift, nWx);
      // proj GEMM + bias + residual; final block writes d_out
      float* po = (li == 2 && bi == 1) ? (float*)d_out : xbuf;
      dim3 g2((unsigned)(C / 32), (unsigned)(Lt / 32));
      gemm_bf16_wmma<<<g2, 32, 0, stream>>>(awbf, pwb[li][bi], pb[li][bi], xbuf,
                                            po, nullptr, (int)Lt, C, C);
    }
    if (li < 2) {
      long total = Bn * (H / 2) * (W / 2) * 4L * C;
      merge_gather_kernel<<<(unsigned)((total + 255) / 256), 256, 0, stream>>>(xbuf, tmpf, H, W, C);
      H /= 2; W /= 2;
      long L2 = Bn * H * W;
      layernorm_kernel<<<(unsigned)L2, 32, 0, stream>>>(tmpf, mng[li], mnb[li],
                                                        nullptr, xnbf, 4 * C);
      dim3 g3((unsigned)(2 * C / 32), (unsigned)(L2 / 32));
      gemm_bf16_wmma<<<g3, 32, 0, stream>>>(xnbf, rwb[li], nullptr, nullptr,
                                            xbuf, nullptr, (int)L2, 2 * C, 4 * C);
    }
  }
}
